// BahdanauDecoder_60370060312744
// MI455X (gfx1250) — matrix operand order repaired and checked by reading, not compiled, not run on hardware
//
#include <hip/hip_runtime.h>

// Problem constants (match reference)
#define N_B  32
#define L_S  128
#define H_D  1024
#define EH_D 1024
#define V_D  32000
#define G3H  3072   // 3*H

typedef __attribute__((__ext_vector_type__(16))) __bf16 v16bf;
typedef __attribute__((__ext_vector_type__(8)))  __bf16 v8bf;
typedef __attribute__((__ext_vector_type__(8)))  float  v8f;

__device__ __forceinline__ v8f wmma_bf16(v16bf a, v16bf b, v8f c) {
  return __builtin_amdgcn_wmma_f32_16x16x32_bf16(false, a, false, b,
                                                 (short)0, c, false, false);
}

// ---------------------------------------------------------------------------
// Skinny GEMM: C[M,Nout] = A[M,K](bf16 row-major) * B[Nout,K]^T (bf16 weights,
// row-major) + bias.  One wave32 computes a 16x64 output slab: 4 N-tiles that
// share a single A fragment (4 independent WMMA chains -> no XDL RAW stalls,
// 4x less A traffic).  Fragment layouts used here (contiguous k, same order for A and B):
//   A 16x32 bf16: lane = m + 16*g ; elems 0..7 -> K=16g+0..7, elems 8..15 ->
//                 K=16g+8..15   (two 16B contiguous loads per lane)
//   B 32x16 bf16: lane = n + 16*g ; elems 0..15 -> K=16g+0..15 (one 32B load)
//   D 16x16 f32 : lane = n + 16*g ; vgpr r -> row = r + 8g, col = n
// Requires: M % 16 == 0, Nout % 64 == 0, K % 32 == 0 (true for all call sites)
// ---------------------------------------------------------------------------
__global__ __launch_bounds__(256)
void gemm_bf16_wmma(const __bf16* __restrict__ A, const __bf16* __restrict__ B,
                    const float* __restrict__ bias, float* __restrict__ C,
                    int M, int Nout, int K) {
  const int wave    = blockIdx.x * (blockDim.x >> 5) + (threadIdx.x >> 5);
  const int tilesN4 = Nout >> 6;      // groups of four 16-wide N tiles
  const int tilesM  = M >> 4;
  if (wave >= tilesM * tilesN4) return;
  const int tm   = wave / tilesN4;
  const int tn0  = (wave - tm * tilesN4) << 2;   // first N-tile of this wave
  const int lane = threadIdx.x & 31;
  const int g    = lane >> 4;     // lane half
  const int mr   = lane & 15;     // row-in-tile for A, col-in-tile for B/D

  const __bf16* __restrict__ arow = A + (size_t)(tm * 16 + mr) * K;
  const __bf16* __restrict__ b0 = B + (size_t)((tn0 + 0) * 16 + mr) * K;
  const __bf16* __restrict__ b1 = B + (size_t)((tn0 + 1) * 16 + mr) * K;
  const __bf16* __restrict__ b2 = B + (size_t)((tn0 + 2) * 16 + mr) * K;
  const __bf16* __restrict__ b3 = B + (size_t)((tn0 + 3) * 16 + mr) * K;

  v8f acc0 = {}, acc1 = {}, acc2 = {}, acc3 = {};
  for (int k0 = 0; k0 < K; k0 += 32) {
    v8bf a_lo = *(const v8bf*)(arow + k0 + 16 * g);
    v8bf a_hi = *(const v8bf*)(arow + k0 + 16 * g + 8);
    v16bf af;
#pragma unroll
    for (int i = 0; i < 8; ++i) { af[i] = a_lo[i]; af[i + 8] = a_hi[i]; }
    v16bf bf0 = *(const v16bf*)(b0 + k0 + 16 * g);
    v16bf bf1 = *(const v16bf*)(b1 + k0 + 16 * g);
    v16bf bf2 = *(const v16bf*)(b2 + k0 + 16 * g);
    v16bf bf3 = *(const v16bf*)(b3 + k0 + 16 * g);
    acc0 = wmma_bf16(af, bf0, acc0);
    acc1 = wmma_bf16(af, bf1, acc1);
    acc2 = wmma_bf16(af, bf2, acc2);
    acc3 = wmma_bf16(af, bf3, acc3);
  }

  float* __restrict__ crow = C + (size_t)(tm * 16 + 8 * g) * Nout + tn0 * 16 + mr;
  const float* bptr = bias ? bias + tn0 * 16 + mr : nullptr;
  const float bv0 = bptr ? bptr[0]  : 0.f;
  const float bv1 = bptr ? bptr[16] : 0.f;
  const float bv2 = bptr ? bptr[32] : 0.f;
  const float bv3 = bptr ? bptr[48] : 0.f;
#pragma unroll
  for (int r = 0; r < 8; ++r) {
    float* __restrict__ cr = crow + (size_t)r * Nout;
    cr[0]  = acc0[r] + bv0;
    cr[16] = acc1[r] + bv1;
    cr[32] = acc2[r] + bv2;
    cr[48] = acc3[r] + bv3;
  }
}

// ---------------------------------------------------------------------------
// Fused attention: scores = Va . tanh(qWa[n] + Uk[n,l]) ; softmax over L;
// context = sum_l w*keys ; writes attention weights to out, builds bf16
// x = [context, embedding[tok]] for the GRU input GEMM.  One block per n.
// ---------------------------------------------------------------------------
__global__ __launch_bounds__(256)
void attn_kernel(const float* __restrict__ qWa, const float* __restrict__ Uk,
                 const __bf16* __restrict__ keys, const float* __restrict__ Va_w,
                 const float* __restrict__ Va_b, const int* __restrict__ tgt,
                 const float* __restrict__ emb, __bf16* __restrict__ x_bf,
                 float* __restrict__ attn_out, int t) {
  __shared__ float s_sc[L_S];
  __shared__ float s_red[256];
  const int n = blockIdx.x, tid = threadIdx.x;
  const int wave = tid >> 5, lane = tid & 31;

  const float* __restrict__ q = qWa + (size_t)n * H_D;
  for (int l = wave; l < L_S; l += 8) {
    const float* __restrict__ u = Uk + (size_t)(n * L_S + l) * H_D;
    float sum = 0.f;
    for (int h = lane; h < H_D; h += 32)
      sum += Va_w[h] * tanhf(q[h] + u[h]);
    for (int off = 16; off > 0; off >>= 1) sum += __shfl_down(sum, off, 32);
    if (lane == 0) s_sc[l] = sum + Va_b[0];
  }
  __syncthreads();

  float m = -1e30f;
  for (int l = tid; l < L_S; l += 256) m = fmaxf(m, s_sc[l]);
  s_red[tid] = m; __syncthreads();
  for (int s = 128; s > 0; s >>= 1) {
    if (tid < s) s_red[tid] = fmaxf(s_red[tid], s_red[tid + s]);
    __syncthreads();
  }
  m = s_red[0]; __syncthreads();

  float e = 0.f;
  for (int l = tid; l < L_S; l += 256) e += __expf(s_sc[l] - m);
  s_red[tid] = e; __syncthreads();
  for (int s = 128; s > 0; s >>= 1) {
    if (tid < s) s_red[tid] += s_red[tid + s];
    __syncthreads();
  }
  const float inv = 1.0f / s_red[0];
  __syncthreads();

  for (int l = tid; l < L_S; l += 256) {
    float w = __expf(s_sc[l] - m) * inv;
    s_sc[l] = w;
    attn_out[(size_t)(n * L_S + t) * L_S + l] = w;
  }
  __syncthreads();

  const int tok = tgt[n * L_S + t];
  for (int h = tid; h < H_D; h += 256) {
    float acc = 0.f;
    const __bf16* __restrict__ kp = keys + (size_t)(n * L_S) * EH_D + h;
#pragma unroll 4
    for (int l = 0; l < L_S; ++l) acc += s_sc[l] * (float)kp[(size_t)l * EH_D];
    x_bf[(size_t)n * 2048 + h]       = (__bf16)acc;
    x_bf[(size_t)n * 2048 + H_D + h] = (__bf16)emb[(size_t)tok * H_D + h];
  }
}

// GRU gate pointwise; keeps fp32 state, also emits bf16 state + concat-q slice.
__global__ __launch_bounds__(256)
void gru_elem_kernel(const float* __restrict__ gi, const float* __restrict__ gh,
                     float* __restrict__ h, __bf16* __restrict__ h_bf,
                     __bf16* __restrict__ q_bf, int qoff) {
  int idx = blockIdx.x * 256 + threadIdx.x;
  if (idx >= N_B * H_D) return;
  int n = idx >> 10, j = idx & 1023;
  const float* __restrict__ gin = gi + (size_t)n * G3H;
  const float* __restrict__ ghn = gh + (size_t)n * G3H;
  float ir = gin[j], iz = gin[1024 + j], in_ = gin[2048 + j];
  float hr = ghn[j], hz = ghn[1024 + j], hn = ghn[2048 + j];
  float r  = 1.f / (1.f + __expf(-(ir + hr)));
  float z  = 1.f / (1.f + __expf(-(iz + hz)));
  float nn = tanhf(in_ + r * hn);
  float hv = (1.f - z) * nn + z * h[idx];
  h[idx] = hv;
  __bf16 hb = (__bf16)hv;
  h_bf[idx] = hb;
  q_bf[(size_t)n * 2048 + qoff + j] = hb;
}

__global__ __launch_bounds__(256)
void log_softmax_kernel(const float* __restrict__ logits,
                        float* __restrict__ out, int t) {
  __shared__ float s_red[256];
  const int n = blockIdx.x, tid = threadIdx.x;
  const float* __restrict__ row = logits + (size_t)n * V_D;
  float m = -1e30f;
  for (int v = tid; v < V_D; v += 256) m = fmaxf(m, row[v]);
  s_red[tid] = m; __syncthreads();
  for (int s = 128; s > 0; s >>= 1) {
    if (tid < s) s_red[tid] = fmaxf(s_red[tid], s_red[tid + s]);
    __syncthreads();
  }
  m = s_red[0]; __syncthreads();
  float e = 0.f;
  for (int v = tid; v < V_D; v += 256) e += __expf(row[v] - m);
  s_red[tid] = e; __syncthreads();
  for (int s = 128; s > 0; s >>= 1) {
    if (tid < s) s_red[tid] += s_red[tid + s];
    __syncthreads();
  }
  const float lse = m + __logf(s_red[0]);
  float* __restrict__ orow = out + (size_t)(n * L_S + t) * V_D;
  for (int v = tid; v < V_D; v += 256) orow[v] = row[v] - lse;
}

__global__ __launch_bounds__(256)
void f32_to_bf16_kernel(const float* __restrict__ s, __bf16* __restrict__ d, int n) {
  int i = blockIdx.x * 256 + threadIdx.x;
  if (i < n) d[i] = (__bf16)s[i];
}

// enc [L,1,N,EH] fp32 -> keys [N,L,EH] bf16
__global__ __launch_bounds__(256)
void build_keys_kernel(const float* __restrict__ enc, __bf16* __restrict__ keys) {
  int i = blockIdx.x * 256 + threadIdx.x;
  if (i >= L_S * N_B * EH_D) return;
  int e = i & 1023, rest = i >> 10;
  int n = rest & 31, l = rest >> 5;
  keys[(size_t)(n * L_S + l) * EH_D + e] = (__bf16)enc[i];
}

// zero fp32 state, bf16 state and the concat-q buffer (bf16 zero == zero bits)
__global__ __launch_bounds__(256)
void init_state_kernel(float* __restrict__ h0, float* __restrict__ h1,
                       __bf16* __restrict__ h0b, __bf16* __restrict__ h1b,
                       __bf16* __restrict__ q) {
  int i = blockIdx.x * 256 + threadIdx.x;
  if (i < N_B * H_D) {
    h0[i] = 0.f; h1[i] = 0.f;
    h0b[i] = (__bf16)0.f; h1b[i] = (__bf16)0.f;
  }
  if (i < N_B * 2048) q[i] = (__bf16)0.f;
}

extern "C" void kernel_launch(void* const* d_in, const int* in_sizes, int n_in,
                              void* d_out, int out_size, void* d_ws, size_t ws_size,
                              hipStream_t stream) {
  const float* enc   = (const float*)d_in[0];
  const int*   tgt   = (const int*)  d_in[1];
  const float* emb   = (const float*)d_in[2];
  const float* Wa_w  = (const float*)d_in[3];
  const float* Wa_b  = (const float*)d_in[4];
  const float* Ua_w  = (const float*)d_in[5];
  const float* Ua_b  = (const float*)d_in[6];
  const float* Va_w  = (const float*)d_in[7];
  const float* Va_b  = (const float*)d_in[8];
  const float* W_ih0 = (const float*)d_in[9];
  const float* W_hh0 = (const float*)d_in[10];
  const float* b_ih0 = (const float*)d_in[11];
  const float* b_hh0 = (const float*)d_in[12];
  const float* W_ih1 = (const float*)d_in[13];
  const float* W_hh1 = (const float*)d_in[14];
  const float* b_ih1 = (const float*)d_in[15];
  const float* b_hh1 = (const float*)d_in[16];
  const float* fc_w  = (const float*)d_in[17];
  const float* fc_b  = (const float*)d_in[18];

  float* dec_out  = (float*)d_out;                               // [N,L,V]
  float* attn_out = dec_out + (size_t)N_B * L_S * V_D;           // [N,L,L]

  // Workspace carve-out (256B aligned slices)
  uint8_t* ws = (uint8_t*)d_ws;
  size_t off = 0;
  auto take = [&](size_t bytes) -> void* {
    void* p = ws + off;
    off += (bytes + 255) & ~(size_t)255;
    return p;
  };
  __bf16* keys_bf = (__bf16*)take((size_t)N_B * L_S * EH_D * 2);
  __bf16* Ua_bf   = (__bf16*)take((size_t)H_D * EH_D * 2);
  __bf16* Wa_bf   = (__bf16*)take((size_t)H_D * 2 * H_D * 2);
  __bf16* Wih0_bf = (__bf16*)take((size_t)G3H * (EH_D + H_D) * 2);
  __bf16* Whh0_bf = (__bf16*)take((size_t)G3H * H_D * 2);
  __bf16* Wih1_bf = (__bf16*)take((size_t)G3H * H_D * 2);
  __bf16* Whh1_bf = (__bf16*)take((size_t)G3H * H_D * 2);
  __bf16* fc_bf   = (__bf16*)take((size_t)V_D * H_D * 2);
  float*  Uk      = (float*) take((size_t)N_B * L_S * H_D * 4);
  float*  qWa     = (float*) take((size_t)N_B * H_D * 4);
  __bf16* q_bf    = (__bf16*)take((size_t)N_B * 2 * H_D * 2);
  __bf16* x_bf    = (__bf16*)take((size_t)N_B * 2 * H_D * 2);
  float*  h0f     = (float*) take((size_t)N_B * H_D * 4);
  float*  h1f     = (float*) take((size_t)N_B * H_D * 4);
  __bf16* h0_bf   = (__bf16*)take((size_t)N_B * H_D * 2);
  __bf16* h1_bf   = (__bf16*)take((size_t)N_B * H_D * 2);
  float*  gi      = (float*) take((size_t)N_B * G3H * 4);
  float*  gh      = (float*) take((size_t)N_B * G3H * 4);
  float*  logits  = (float*) take((size_t)N_B * V_D * 4);

  auto cvt = [&](const float* s, __bf16* d, size_t n) {
    f32_to_bf16_kernel<<<(int)((n + 255) / 256), 256, 0, stream>>>(s, d, (int)n);
  };
  auto gemm = [&](const __bf16* A, const __bf16* B, const float* bias, float* C,
                  int M, int Nout, int K) {
    int waves = (M >> 4) * (Nout >> 6);   // one wave per 16x64 slab
    gemm_bf16_wmma<<<(waves + 7) / 8, 256, 0, stream>>>(A, B, bias, C, M, Nout, K);
  };

  // ---- one-time precompute: bf16 weights, keys, Uk = keys @ Ua^T + b ----
  cvt(Ua_w,  Ua_bf,   (size_t)H_D * EH_D);
  cvt(Wa_w,  Wa_bf,   (size_t)H_D * 2 * H_D);
  cvt(W_ih0, Wih0_bf, (size_t)G3H * (EH_D + H_D));
  cvt(W_hh0, Whh0_bf, (size_t)G3H * H_D);
  cvt(W_ih1, Wih1_bf, (size_t)G3H * H_D);
  cvt(W_hh1, Whh1_bf, (size_t)G3H * H_D);
  cvt(fc_w,  fc_bf,   (size_t)V_D * H_D);
  build_keys_kernel<<<(L_S * N_B * EH_D) / 256, 256, 0, stream>>>(enc, keys_bf);
  gemm(keys_bf, Ua_bf, Ua_b, Uk, N_B * L_S, H_D, EH_D);
  init_state_kernel<<<(N_B * 2 * H_D) / 256, 256, 0, stream>>>(h0f, h1f, h0_bf,
                                                               h1_bf, q_bf);

  // ---- sequential decode over L steps ----
  for (int t = 0; t < L_S; ++t) {
    gemm(q_bf, Wa_bf, Wa_b, qWa, N_B, H_D, 2 * H_D);
    attn_kernel<<<N_B, 256, 0, stream>>>(qWa, Uk, keys_bf, Va_w, Va_b, tgt, emb,
                                         x_bf, attn_out, t);
    // GRU layer 0
    gemm(x_bf,  Wih0_bf, b_ih0, gi, N_B, G3H, EH_D + H_D);
    gemm(h0_bf, Whh0_bf, b_hh0, gh, N_B, G3H, H_D);
    gru_elem_kernel<<<(N_B * H_D) / 256, 256, 0, stream>>>(gi, gh, h0f, h0_bf,
                                                           q_bf, 0);
    // GRU layer 1
    gemm(h0_bf, Wih1_bf, b_ih1, gi, N_B, G3H, H_D);
    gemm(h1_bf, Whh1_bf, b_hh1, gh, N_B, G3H, H_D);
    gru_elem_kernel<<<(N_B * H_D) / 256, 256, 0, stream>>>(gi, gh, h1f, h1_bf,
                                                           q_bf, H_D);
    // classifier + log-softmax
    gemm(h1_bf, fc_bf, fc_b, logits, N_B, V_D, H_D);
    log_softmax_kernel<<<N_B, 256, 0, stream>>>(logits, dec_out, t);
  }
}
